// CausalAttention_89816356094699
// MI455X (gfx1250) — matrix operand order repaired and checked by reading, not compiled, not run on hardware
//
#include <hip/hip_runtime.h>

typedef __attribute__((ext_vector_type(16))) __bf16 v16bf;
typedef __attribute__((ext_vector_type(8)))  __bf16 v8bf;
typedef __attribute__((ext_vector_type(8)))  float  v8f;
typedef __attribute__((ext_vector_type(4)))  unsigned int v4u;
typedef __attribute__((ext_vector_type(8)))  int v8i;
typedef __attribute__((ext_vector_type(4)))  int v4i;
typedef __bf16 bf16;

#define D_MODEL 1024
#define SEQ     2048
#define NBATCH  4
#define SCALE   0.03125f   /* 1/sqrt(1024) */
#define BROW    136        /* 128 k-elements + 8 elem (16B) LDS row padding */

// ---------- WMMA fragment helpers (CDNA5 bf16 16x16x32 layouts) ----------
__device__ __forceinline__ v16bf load_a_frag(const bf16* tile, int ld, int lane) {
    const int m  = lane & 15;
    const int hi = lane >> 4;
    const bf16* row = tile + (size_t)m * ld;
    v8bf lo = *(const v8bf*)(row + hi * 16);
    v8bf hv = *(const v8bf*)(row + hi * 16 + 8);
    v16bf f;
#pragma unroll
    for (int i = 0; i < 8; ++i) { f[i] = lo[i]; f[i + 8] = hv[i]; }
    return f;
}

__device__ __forceinline__ v16bf load_b_frag(const bf16* btile, int ld, int lane) {
    const int n  = lane & 15;
    const int hi = lane >> 4;
    return *(const v16bf*)(btile + (size_t)n * ld + hi * 16);
}

__device__ __forceinline__ v8f wmma_bf16(v16bf a, v16bf b, v8f c) {
    return __builtin_amdgcn_wmma_f32_16x16x32_bf16(false, a, false, b, (short)0, c, false, false);
}

// ---------- Tensor Data Mover: 2-D bf16 tile -> LDS, 16B pad per 256B row ----------
__device__ __forceinline__ void tdm_load_2d(unsigned lds_off, const void* gaddr,
                                            unsigned tile_d0, unsigned tile_d1,
                                            unsigned tensor_d0, unsigned tensor_d1,
                                            unsigned long long stride0_elems) {
    unsigned long long ga = (unsigned long long)gaddr;
    v4u g0;
    g0[0] = 1u;                                                // count=1, user mode
    g0[1] = lds_off;                                           // lds_addr (bytes)
    g0[2] = (unsigned)(ga & 0xFFFFFFFFu);                      // global_addr[31:0]
    g0[3] = (unsigned)((ga >> 32) & 0x01FFFFFFu) | (2u << 30); // addr[56:32] | type=2
    v8i g1;
    g1[0] = (int)((1u << 16)                                   // data_size=1 (2 bytes)
                | (1u << 20)                                   // pad_enable
                | (5u << 22)                                   // pad_interval: 64 DWORDs
                | (3u << 25));                                 // pad_amount: 4 DWORDs
    g1[1] = (int)((tensor_d0 & 0xFFFFu) << 16);
    g1[2] = (int)(((tensor_d0 >> 16) & 0xFFFFu) | ((tensor_d1 & 0xFFFFu) << 16));
    g1[3] = (int)(((tensor_d1 >> 16) & 0xFFFFu) | ((tile_d0 & 0xFFFFu) << 16));
    g1[4] = (int)(tile_d1 & 0xFFFFu);
    g1[5] = (int)(unsigned)(stride0_elems & 0xFFFFFFFFu);
    g1[6] = (int)(unsigned)((stride0_elems >> 32) & 0xFFFFu);
    g1[7] = 0;
    v4i z4 = {0, 0, 0, 0};
#if defined(__clang_major__) && (__clang_major__ >= 23)
    v8i z8 = {0, 0, 0, 0, 0, 0, 0, 0};
    __builtin_amdgcn_tensor_load_to_lds(g0, g1, z4, z4, z8, 0);
#else
    __builtin_amdgcn_tensor_load_to_lds(g0, g1, z4, z4, 0);
#endif
}

// ---------- cast kernels ----------
__global__ void cast_x_kernel(const float* __restrict__ x, bf16* __restrict__ xb, int n) {
    int i = blockIdx.x * blockDim.x + threadIdx.x;
    int stride = gridDim.x * blockDim.x;
    for (; i < n; i += stride) xb[i] = (bf16)x[i];
}

__global__ void cast_wt_kernel(const float* __restrict__ wq, const float* __restrict__ wk,
                               const float* __restrict__ wv, bf16* __restrict__ wt) {
    int i = blockIdx.x * blockDim.x + threadIdx.x;
    if (i >= 3 * 1024 * 1024) return;
    int w = i >> 20;
    int r = i & ((1 << 20) - 1);
    int n = r >> 10;
    int k = r & 1023;
    const float* src = (w == 0) ? wq : (w == 1) ? wk : wv;
    wt[i] = (bf16)src[k * D_MODEL + n];
}

// ---------- QKV projection: TDM-staged W^T panel in LDS, 16x64 per wave ----------
__global__ void proj_kernel(const bf16* __restrict__ xb, const bf16* __restrict__ wt,
                            bf16* __restrict__ qb, bf16* __restrict__ kb, bf16* __restrict__ vt) {
    __shared__ bf16 ldsB[2][64 * BROW];
    const int lane  = threadIdx.x & 31;
    const int wave  = threadIdx.x >> 5;
    const int n0    = blockIdx.x * 64;
    const int mTile = blockIdx.y * 4 + wave;
    const int which = blockIdx.z;
    const bf16* a_base = xb + (size_t)mTile * 16 * D_MODEL;
    const bf16* wpanel = wt + (size_t)which * D_MODEL * D_MODEL + (size_t)n0 * D_MODEL;

    v8f acc[4] = {};
    if (wave == 0)
        tdm_load_2d((unsigned)(uintptr_t)&ldsB[0][0], wpanel, 128, 64, 1024, 64, 1024ull);
    for (int c = 0; c < 8; ++c) {
        const int buf = c & 1;
        if (wave == 0) {
            if (c + 1 < 8) {
                tdm_load_2d((unsigned)(uintptr_t)&ldsB[buf ^ 1][0], wpanel + (c + 1) * 128,
                            128, 64, 1024, 64, 1024ull);
                __builtin_amdgcn_s_wait_tensorcnt((short)1);
            } else {
                __builtin_amdgcn_s_wait_tensorcnt((short)0);
            }
        }
        __syncthreads();
#pragma unroll
        for (int kc = 0; kc < 128; kc += 32) {
            v16bf a = load_a_frag(a_base + c * 128 + kc, D_MODEL, lane);
#pragma unroll
            for (int j = 0; j < 4; ++j) {
                v16bf b = load_b_frag(&ldsB[buf][j * 16 * BROW + kc], BROW, lane);
                acc[j] = wmma_bf16(a, b, acc[j]);
            }
        }
        __syncthreads();
    }
    const int n     = lane & 15;
    const int mbase = (lane >> 4) * 8;
    if (which < 2) {
        bf16* base = (which == 0 ? qb : kb) + ((size_t)mTile * 16 + mbase) * D_MODEL + n0 + n;
#pragma unroll
        for (int j = 0; j < 4; ++j)
#pragma unroll
            for (int g = 0; g < 8; ++g)
                base[(size_t)g * D_MODEL + j * 16] = (bf16)acc[j][g];
    } else {
        int mglob = mTile * 16;
        int bidx  = mglob / SEQ;
        int t0    = mglob - bidx * SEQ + mbase;
#pragma unroll
        for (int j = 0; j < 4; ++j) {
            int nglob = n0 + j * 16 + n;
            v8bf o;
#pragma unroll
            for (int g = 0; g < 8; ++g) o[g] = (bf16)acc[j][g];
            *(v8bf*)(vt + (size_t)bidx * D_MODEL * SEQ + (size_t)nglob * SEQ + t0) = o;
        }
    }
}

// ---------- pass 1: online softmax stats; 8 q-tiles/block share TDM-staged K panel ----------
__global__ void stats_kernel(const bf16* __restrict__ qb, const bf16* __restrict__ kb,
                             float* __restrict__ marr, float* __restrict__ larr) {
    __shared__ bf16 ldsB[2][64 * BROW];
    const int lane  = threadIdx.x & 31;
    const int wave  = threadIdx.x >> 5;
    const int qt    = blockIdx.x * 8 + wave;   // this wave's q-tile
    const int qtMax = blockIdx.x * 8 + 7;      // block-uniform loop bound
    const int bat   = blockIdx.y;
    const bf16* qbase = qb + ((size_t)bat * SEQ + qt * 16) * D_MODEL;
    const bf16* kbb   = kb + (size_t)bat * SEQ * D_MODEL;
    const int n     = lane & 15;
    const int mbase = (lane >> 4) * 8;
    float m8[8], l8[8];
#pragma unroll
    for (int g = 0; g < 8; ++g) { m8[g] = -__builtin_inff(); l8[g] = 0.f; }

    for (int kt4 = 0; kt4 <= qtMax; kt4 += 4) {          // 4 key tiles / group
        const bf16* kpanel = kbb + (size_t)kt4 * 16 * D_MODEL;   // 64 K rows
        v8f acc[4] = {};
        for (int c = 0; c < 8; ++c) {                    // 8 k-chunks of 128
            const int buf = c & 1;
            if (wave == 0) {
                if (c == 0)
                    tdm_load_2d((unsigned)(uintptr_t)&ldsB[0][0], kpanel, 128, 64,
                                1024, 64, 1024ull);
                if (c + 1 < 8) {
                    tdm_load_2d((unsigned)(uintptr_t)&ldsB[buf ^ 1][0], kpanel + (c + 1) * 128,
                                128, 64, 1024, 64, 1024ull);
                    __builtin_amdgcn_s_wait_tensorcnt((short)1);
                } else {
                    __builtin_amdgcn_s_wait_tensorcnt((short)0);
                }
            }
            __syncthreads();
            if (qt >= kt4) {                             // wave still inside causal range
#pragma unroll
                for (int kc = 0; kc < 128; kc += 32) {
                    v16bf a = load_a_frag(qbase + c * 128 + kc, D_MODEL, lane);
#pragma unroll
                    for (int j = 0; j < 4; ++j) {
                        v16bf bfr = load_b_frag(&ldsB[buf][j * 16 * BROW + kc], BROW, lane);
                        acc[j] = wmma_bf16(a, bfr, acc[j]);
                    }
                }
            }
            __syncthreads();
        }
        if (qt >= kt4) {
#pragma unroll
            for (int j = 0; j < 4; ++j) {
                const int kIdx = (kt4 + j) * 16 + n;
#pragma unroll
                for (int g = 0; g < 8; ++g) {
                    const int qIdx = qt * 16 + mbase + g;
                    float s = (kIdx <= qIdx) ? acc[j][g] * SCALE : -__builtin_inff();
                    float tmax = s;
                    tmax = fmaxf(tmax, __shfl_xor(tmax, 1, 16));
                    tmax = fmaxf(tmax, __shfl_xor(tmax, 2, 16));
                    tmax = fmaxf(tmax, __shfl_xor(tmax, 4, 16));
                    tmax = fmaxf(tmax, __shfl_xor(tmax, 8, 16));
                    float mnew = fmaxf(m8[g], tmax);
                    float p = __expf(s - mnew);
                    p += __shfl_xor(p, 1, 16);
                    p += __shfl_xor(p, 2, 16);
                    p += __shfl_xor(p, 4, 16);
                    p += __shfl_xor(p, 8, 16);
                    l8[g] = l8[g] * __expf(m8[g] - mnew) + p;
                    m8[g] = mnew;
                }
            }
        }
    }
    if (n == 0) {
        int row = bat * SEQ + qt * 16 + mbase;
#pragma unroll
        for (int g = 0; g < 8; ++g) { marr[row + g] = m8[g]; larr[row + g] = l8[g]; }
    }
}

// ---------- pass 2: weights P; 8 q-tiles/block share TDM-staged K panel ----------
__global__ void pmat_kernel(const bf16* __restrict__ qb, const bf16* __restrict__ kb,
                            const float* __restrict__ marr, const float* __restrict__ larr,
                            bf16* __restrict__ p) {
    __shared__ bf16 ldsB[2][64 * BROW];
    const int lane  = threadIdx.x & 31;
    const int wave  = threadIdx.x >> 5;
    const int ktB   = blockIdx.x * 4;          // 4 key tiles per block (grid.x = 32)
    const int qt    = blockIdx.y * 8 + wave;   // grid.y = 16
    const int qtMax = blockIdx.y * 8 + 7;
    const int bat   = blockIdx.z;
    const int n     = lane & 15;
    const int mbase = (lane >> 4) * 8;
    bf16* prow = p + ((size_t)bat * SEQ + qt * 16 + mbase) * SEQ + ktB * 16 + n;
    if (ktB > qtMax) {   // whole block above the diagonal: zero-fill, no barriers
#pragma unroll
        for (int j = 0; j < 4; ++j)
#pragma unroll
            for (int g = 0; g < 8; ++g) prow[(size_t)g * SEQ + j * 16] = (bf16)0.f;
        return;
    }
    v8f acc[4] = {};
    const bf16* qbase  = qb + ((size_t)bat * SEQ + qt * 16) * D_MODEL;
    const bf16* kpanel = kb + ((size_t)bat * SEQ + ktB * 16) * D_MODEL;
    if (wave == 0)
        tdm_load_2d((unsigned)(uintptr_t)&ldsB[0][0], kpanel, 128, 64, 1024, 64, 1024ull);
    for (int c = 0; c < 8; ++c) {
        const int buf = c & 1;
        if (wave == 0) {
            if (c + 1 < 8) {
                tdm_load_2d((unsigned)(uintptr_t)&ldsB[buf ^ 1][0], kpanel + (c + 1) * 128,
                            128, 64, 1024, 64, 1024ull);
                __builtin_amdgcn_s_wait_tensorcnt((short)1);
            } else {
                __builtin_amdgcn_s_wait_tensorcnt((short)0);
            }
        }
        __syncthreads();
        if (qt >= ktB) {
#pragma unroll
            for (int kc = 0; kc < 128; kc += 32) {
                v16bf a = load_a_frag(qbase + c * 128 + kc, D_MODEL, lane);
#pragma unroll
                for (int j = 0; j < 4; ++j) {
                    v16bf bfr = load_b_frag(&ldsB[buf][j * 16 * BROW + kc], BROW, lane);
                    acc[j] = wmma_bf16(a, bfr, acc[j]);
                }
            }
        }
        __syncthreads();
    }
#pragma unroll
    for (int g = 0; g < 8; ++g) {
        int qIdx   = qt * 16 + mbase + g;
        float mrow = marr[(size_t)bat * SEQ + qIdx];
        float inv  = 1.0f / larr[(size_t)bat * SEQ + qIdx];
#pragma unroll
        for (int j = 0; j < 4; ++j) {
            int kIdx = (ktB + j) * 16 + n;
            float v  = (kIdx <= qIdx) ? __expf(acc[j][g] * SCALE - mrow) * inv : 0.f;
            prow[(size_t)g * SEQ + j * 16] = (bf16)v;
        }
    }
}

// ---------- O = P @ V; 8 q-tiles/block share TDM-staged V^T panel ----------
__global__ void out_kernel(const bf16* __restrict__ p, const bf16* __restrict__ vt,
                           float* __restrict__ out) {
    __shared__ bf16 ldsB[2][64 * BROW];
    const int lane  = threadIdx.x & 31;
    const int wave  = threadIdx.x >> 5;
    const int nB    = blockIdx.x * 64;         // grid.x = 16
    const int qt    = blockIdx.y * 8 + wave;   // grid.y = 16
    const int qtMax = blockIdx.y * 8 + 7;
    const int bat   = blockIdx.z;
    const bf16* abase = p  + ((size_t)bat * SEQ + qt * 16) * SEQ;
    const bf16* bbase = vt + (size_t)bat * D_MODEL * SEQ + (size_t)nB * SEQ;
    const int kEnd    = ((qt + 1) * 16 + 31) & ~31;        // P is zero-padded past causal
    const int nchunks = (((qtMax + 1) * 16 + 127) & ~127) >> 7;
    v8f acc[4] = {};
    if (wave == 0)
        tdm_load_2d((unsigned)(uintptr_t)&ldsB[0][0], bbase, 128, 64, 2048, 64, 2048ull);
    for (int c = 0; c < nchunks; ++c) {
        const int buf = c & 1;
        if (wave == 0) {
            if (c + 1 < nchunks) {
                tdm_load_2d((unsigned)(uintptr_t)&ldsB[buf ^ 1][0], bbase + (c + 1) * 128,
                            128, 64, 2048, 64, 2048ull);
                __builtin_amdgcn_s_wait_tensorcnt((short)1);
            } else {
                __builtin_amdgcn_s_wait_tensorcnt((short)0);
            }
        }
        __syncthreads();
#pragma unroll
        for (int kc = 0; kc < 128; kc += 32) {
            const int kk = c * 128 + kc;
            if (kk < kEnd) {                  // wave-uniform causal truncation
                v16bf a = load_a_frag(abase + kk, SEQ, lane);
#pragma unroll
                for (int j = 0; j < 4; ++j) {
                    v16bf bfr = load_b_frag(&ldsB[buf][j * 16 * BROW + kc], BROW, lane);
                    acc[j] = wmma_bf16(a, bfr, acc[j]);
                }
            }
        }
        __syncthreads();
    }
    const int n     = lane & 15;
    const int mbase = (lane >> 4) * 8;
    float* dst = out + ((size_t)bat * SEQ + qt * 16 + mbase) * D_MODEL + nB + n;
#pragma unroll
    for (int j = 0; j < 4; ++j)
#pragma unroll
        for (int g = 0; g < 8; ++g)
            dst[(size_t)g * D_MODEL + j * 16] = acc[j][g];
}

extern "C" void kernel_launch(void* const* d_in, const int* in_sizes, int n_in,
                              void* d_out, int out_size, void* d_ws, size_t ws_size,
                              hipStream_t stream) {
    const float* x  = (const float*)d_in[0];
    const float* wq = (const float*)d_in[1];
    const float* wk = (const float*)d_in[2];
    const float* wv = (const float*)d_in[3];
    float* out = (float*)d_out;

    char* ws = (char*)d_ws;
    size_t off = 0;
    auto alloc = [&](size_t bytes) -> void* {
        void* pp = ws + off;
        off = (off + bytes + 255) & ~(size_t)255;
        return pp;
    };
    const size_t MT = (size_t)NBATCH * SEQ;
    bf16*  xb   = (bf16*)alloc(MT * D_MODEL * 2);
    bf16*  wt   = (bf16*)alloc(3ull * D_MODEL * D_MODEL * 2);
    bf16*  qb   = (bf16*)alloc(MT * D_MODEL * 2);
    bf16*  kb   = (bf16*)alloc(MT * D_MODEL * 2);
    bf16*  vt   = (bf16*)alloc(MT * D_MODEL * 2);
    float* marr = (float*)alloc(MT * 4);
    float* larr = (float*)alloc(MT * 4);
    bf16*  pmat = (bf16*)alloc((size_t)NBATCH * SEQ * SEQ * 2);

    cast_x_kernel <<<dim3(4096),           dim3(256), 0, stream>>>(x, xb, (int)(MT * D_MODEL));
    cast_wt_kernel<<<dim3(12288),          dim3(256), 0, stream>>>(wq, wk, wv, wt); // 3*2^20/256
    proj_kernel   <<<dim3(16, 128, 3),     dim3(128), 0, stream>>>(xb, wt, qb, kb, vt);
    stats_kernel  <<<dim3(16, NBATCH),     dim3(256), 0, stream>>>(qb, kb, marr, larr);
    pmat_kernel   <<<dim3(32, 16, NBATCH), dim3(256), 0, stream>>>(qb, kb, marr, larr, pmat);
    out_kernel    <<<dim3(16, 16, NBATCH), dim3(256), 0, stream>>>(pmat, vt, out);
}
